// MultiTaskLoss_62105227100175
// MI455X (gfx1250) — compile-verified, matrix operand order checked by reading, not run on hardware
//
#include <hip/hip_runtime.h>
#include <hip/hip_bf16.h>
#include <math.h>

// Problem dims (fixed by setup_inputs)
#define NB   64
#define HH   512
#define WWID 512
#define NPIX (HH * WWID)
#define WROW (WWID / 32)   // 16 bitmask words per row
#define TILE 64
#define HALO 10
#define TROWS (TILE + 2 * HALO)  // 84
#define TWORDS 4                 // 128 bits per LDS row covers tile + halo

typedef __attribute__((ext_vector_type(2)))  float    v2f;
typedef __attribute__((ext_vector_type(8)))  float    v8f;
typedef __attribute__((ext_vector_type(16))) _Float16 v16h;

// ---- feature detection (per compile pass; host pass safely falls back) ----
#if __has_builtin(__builtin_amdgcn_wmma_f32_16x16x4_f32)
#define KB_WMMA32 1
#elif __has_builtin(__builtin_amdgcn_wmma_f32_16x16x32_f16)
#define KB_WMMA16 1
#endif

#if __has_builtin(__builtin_amdgcn_global_load_async_to_lds_b32)
#define KB_ASYNC 1
#endif

__device__ __forceinline__ float wave_sum(float v) {
#if __has_builtin(__builtin_amdgcn_ds_swizzle)
  // group-of-32 XOR swizzles: imm = (xor_mask<<10) | (or=0<<5) | (and=0x1F)
  v += __int_as_float(__builtin_amdgcn_ds_swizzle(__float_as_int(v), 0x401F));
  v += __int_as_float(__builtin_amdgcn_ds_swizzle(__float_as_int(v), 0x201F));
  v += __int_as_float(__builtin_amdgcn_ds_swizzle(__float_as_int(v), 0x101F));
  v += __int_as_float(__builtin_amdgcn_ds_swizzle(__float_as_int(v), 0x081F));
  v += __int_as_float(__builtin_amdgcn_ds_swizzle(__float_as_int(v), 0x041F));
#else
#pragma unroll
  for (int m = 16; m >= 1; m >>= 1) v += __shfl_xor(v, m, 32);
#endif
  return v;
}

__device__ __forceinline__ unsigned win_extract(unsigned hi, unsigned lo, int off) {
#if __has_builtin(__builtin_amdgcn_alignbit)
  return __builtin_amdgcn_alignbit(hi, lo, off);
#else
  return (unsigned)(((((unsigned long long)hi) << 32) | lo) >> off);
#endif
}

// ---------------------------------------------------------------------------
// K0: zero the 64x8 accumulator table.
// ---------------------------------------------------------------------------
__global__ void k_zero(float* __restrict__ acc) {
  acc[threadIdx.x] = 0.0f;  // <<<1, 512>>>, 64 samples * 8 slots
}

// ---------------------------------------------------------------------------
// K1: per-sample elementwise sums: inter, sum_pc, sum_t, bce_logits,
//     fg_count, bce_fg, bce_bg   -> acc[b*8 + 0..6]
//     Single expf+log1pf per pixel: with L = log1p(e^{-|x|}),
//       sigmoid(x)     = (x>=0 ? 1 : e^{-|x|}) / (1 + e^{-|x|})
//       bce_logits     = max(x,0) - x*t + L
//       log(pc)        = clamp(-(max(-x,0)+L), log 1e-7, log(1-1e-7))
//       log(1-pc)      = clamp(-(max( x,0)+L), log 1e-7, log(1-1e-7))
//     Also bitpacks targets (t > 0.5) via wave32 ballot (fused pass).
// ---------------------------------------------------------------------------
__global__ void k_sums(const float* __restrict__ pred,
                       const float* __restrict__ tgt,
                       float* __restrict__ acc,
                       unsigned* __restrict__ bits) {
  const int b = blockIdx.x >> 5;
  const int chunk = blockIdx.x & 31;
  const float* xp = pred + (size_t)b * NPIX;
  const float* tp = tgt + (size_t)b * NPIX;
  const size_t bitbase = (size_t)b * (NPIX / 32);
  const float C0 = logf(1e-7f);           // const-folded
  const float C1 = logf(1.0f - 1e-7f);    // const-folded
  int base = chunk * (256 * 32) + threadIdx.x;
  float s_inter = 0.f, s_pc = 0.f, s_t = 0.f, s_bce = 0.f;
  float s_fg = 0.f, s_bfg = 0.f, s_bbg = 0.f;
#pragma unroll 4
  for (int i = 0; i < 32; ++i) {
    int p = base + i * 256;
    float x = xp[p];
    float t = tp[p];
    float E = expf(-fabsf(x));
    float L = log1pf(E);
    float sig = ((x >= 0.0f) ? 1.0f : E) / (1.0f + E);
    float pc = fminf(fmaxf(sig, 1e-7f), 1.0f - 1e-7f);
    s_inter += pc * t;
    s_pc += pc;
    s_t += t;
    s_bce += fmaxf(x, 0.0f) - x * t + L;
    float tb = (t > 0.5f) ? 1.0f : 0.0f;
    float lp = fminf(fmaxf(-(fmaxf(-x, 0.0f) + L), C0), C1);  // log(pc)
    float lq = fminf(fmaxf(-(fmaxf(x, 0.0f) + L), C0), C1);   // log(1-pc)
    float bp = -(t * lp + (1.0f - t) * lq);
    s_fg += tb;
    s_bfg += tb * bp;
    s_bbg += (1.0f - tb) * bp;
    // bitpack this 32-px group (lanes are consecutive pixels, 32-aligned)
    unsigned mword;
#if __has_builtin(__builtin_amdgcn_ballot_w32)
    mword = __builtin_amdgcn_ballot_w32(t > 0.5f);
#else
    mword = (unsigned)__ballot(t > 0.5f);
#endif
    if ((threadIdx.x & 31) == 0) bits[bitbase + (unsigned)(p >> 5)] = mword;
  }
  s_inter = wave_sum(s_inter);
  s_pc    = wave_sum(s_pc);
  s_t     = wave_sum(s_t);
  s_bce   = wave_sum(s_bce);
  s_fg    = wave_sum(s_fg);
  s_bfg   = wave_sum(s_bfg);
  s_bbg   = wave_sum(s_bbg);
  if ((threadIdx.x & 31) == 0) {
    float* a = acc + b * 8;
    atomicAdd(a + 0, s_inter);
    atomicAdd(a + 1, s_pc);
    atomicAdd(a + 2, s_t);
    atomicAdd(a + 3, s_bce);
    atomicAdd(a + 4, s_fg);
    atomicAdd(a + 5, s_bfg);
    atomicAdd(a + 6, s_bbg);
  }
}

// ---------------------------------------------------------------------------
// K2: exact L2 distance transform (clamped at 10) + boundary loss partials.
//     One block = one 64x64 tile. Halo bitmask tile staged in LDS via
//     async-to-LDS. Nearest-set-bit squared distance mdx^2 is computed ONCE
//     per (row, column) -- branchless via sentinel bits -- and reused by all
//     16 pixels of the thread (md2[36] in registers; per pixel 21x add+umin).
//     The clamp at d2=100 subsumes both the circular mask and the no-hit
//     case. Boundary term reduces to dist*pc (fg pixels have dist==0).
//     Block reduction of the partials goes through WMMA (B = ones).
// ---------------------------------------------------------------------------
__global__ void k_boundary(const float* __restrict__ pred,
                           const unsigned* __restrict__ bits,
                           float* __restrict__ acc) {
  __shared__ __align__(16) unsigned tile[TROWS][TWORDS];

  const int bid = blockIdx.x;
  const int b = bid >> 6;           // 64 tiles (8x8) per sample
  const int t6 = bid & 63;
  const int tileY = (t6 >> 3) * TILE;
  const int tileX = (t6 & 7) * TILE;
  const unsigned* mrow = bits + (size_t)b * HH * WROW;
  const int wX0 = (tileX >> 5) - 1;  // leftmost LDS word (may be -1 => bg)

  // ---- stage halo bitmask tile into LDS ----
  for (int i = threadIdx.x; i < TROWS * TWORDS; i += 256) {
    int r = i / TWORDS, c = i % TWORDS;
    int gy = tileY - HALO + r;
    int gw = wX0 + c;
    bool ok = (gy >= 0) && (gy < HH) && (gw >= 0) && (gw < WROW);
#if KB_ASYNC
    if (ok) {
      __builtin_amdgcn_global_load_async_to_lds_b32(
          (__attribute__((address_space(1))) int*)(mrow + gy * WROW + gw),
          (__attribute__((address_space(3))) int*)&tile[r][c],
          0, 0);
    } else {
      tile[r][c] = 0u;
    }
#else
    tile[r][c] = ok ? mrow[gy * WROW + gw] : 0u;
#endif
  }
#if KB_ASYNC
#if __has_builtin(__builtin_amdgcn_s_wait_asynccnt)
  __builtin_amdgcn_s_wait_asynccnt(0);
#else
  asm volatile("s_wait_asynccnt 0" ::: "memory");
#endif
#endif
  __syncthreads();

  const int lx = threadIdx.x & 63;   // x within tile (fixed per thread)
  const int yg = threadIdx.x >> 6;   // 0..3, each does 16 consecutive rows
  const int gx = tileX + lx;
  const int s = 22 + lx;             // window start bit = (32+lx) - 10
  const int wsel = s >> 5;           // 0..2 (fixed per thread)
  const int off = s & 31;
  const float* xp = pred + (size_t)b * NPIX;

  // Per-thread base pointer: word `wsel` of tile row yg*16. Row r is at a
  // compile-time dword offset 4*r from here -> ds_load_2addr_b32, no branches.
  const unsigned* rowp = &tile[yg * 16][0] + wsel;

  // ---- per-row nearest-bit squared distance (36 rows per thread) ----
  // Sentinels: right|0x800 -> ctz<=11; left|1 -> dL<=10. A spurious value
  // always yields d2 >= 100, which the clamp maps to dist=10 (= reference).
  unsigned md2[36];
#pragma unroll
  for (int r = 0; r < 36; ++r) {
    unsigned lo = rowp[r * TWORDS];
    unsigned hi = rowp[r * TWORDS + 1];
    unsigned win = win_extract(hi, lo, off) & 0x1FFFFFu;  // 21-bit window
    unsigned right = (win >> 10) | 0x800u;  // dx >= 0 (bit 10 = center)
    unsigned left = (win & 0x7FFu) | 1u;    // dx <= 0
    int dR = __builtin_ctz(right);
    int dL = __builtin_clz(left) - 21;      // 10 - msb(left)
    unsigned mdx = (unsigned)(dR < dL ? dR : dL);
    md2[r] = mdx * mdx;
  }

  // ---- per-pixel: min over 21 rows of dy^2 + mdx^2, then boundary term ----
  float bsum0 = 0.f, bsum1 = 0.f;
#pragma unroll
  for (int i = 0; i < 16; ++i) {
    const int gy = tileY + yg * 16 + i;
    unsigned m = 100u;                 // clamp: dist <= 10
#pragma unroll
    for (int dy = -10; dy <= 10; ++dy) {
      unsigned d2 = (unsigned)(dy * dy) + md2[i + dy + 10];
      m = (d2 < m) ? d2 : m;
    }
    float dist = sqrtf((float)m);

    float x = xp[gy * WWID + gx];
    float E = __expf(-fabsf(x));
    float sig = ((x >= 0.0f) ? 1.0f : E) / (1.0f + E);
    float pc = fminf(fmaxf(sig, 1e-7f), 1.0f - 1e-7f);
    // fg pixel => dist==0 => term==0; bg pixel => |pc-tb| == pc
    float term = dist * pc;
    if (i & 1) bsum1 += term; else bsum0 += term;
  }

  // ---- block reduction of (bsum0, bsum1): WMMA with ones B-matrix ----
  // All 256 threads active here (EXEC all ones, required by WMMA).
  float tot;
#if KB_WMMA32
  {
    v2f a; a.x = bsum0; a.y = bsum1;
    v2f bones; bones.x = 1.0f; bones.y = 1.0f;
    v8f c = {};
    // D[m][n] = sum_k A[m][k] * 1  -> sum over all C = 16 * sum over all A
    c = __builtin_amdgcn_wmma_f32_16x16x4_f32(false, a, false, bones,
                                              (short)0, c, false, false);
    tot = c[0] + c[1] + c[2] + c[3] + c[4] + c[5] + c[6] + c[7];
    tot = wave_sum(tot) * (1.0f / 16.0f);
  }
#elif KB_WMMA16
  {
    v16h a = {};
    a[0] = (_Float16)bsum0;
    a[1] = (_Float16)bsum1;
    v16h bones;
#pragma unroll
    for (int j = 0; j < 16; ++j) bones[j] = (_Float16)1.0f;
    v8f c = {};
    c = __builtin_amdgcn_wmma_f32_16x16x32_f16(false, a, false, bones,
                                               (short)0, c, false, false);
    tot = c[0] + c[1] + c[2] + c[3] + c[4] + c[5] + c[6] + c[7];
    tot = wave_sum(tot) * (1.0f / 16.0f);
  }
#else
  tot = wave_sum(bsum0 + bsum1);
#endif
  if ((threadIdx.x & 31) == 0) atomicAdd(&acc[b * 8 + 7], tot);
}

// ---------------------------------------------------------------------------
// K3: combine per-sample sums into final scalar mean loss.
// ---------------------------------------------------------------------------
__global__ void k_final(const float* __restrict__ acc,
                        const int* __restrict__ task,
                        float* __restrict__ out) {
  __shared__ float red[NB];
  int b = threadIdx.x;  // 64 threads
  const float* a = acc + b * 8;
  float inter = a[0], spc = a[1], st = a[2], bce = a[3];
  float fg = a[4], bfg = a[5], bbg = a[6], bnd = a[7];
  const float N = (float)NPIX;

  float dice = 1.0f - (2.0f * inter + 1e-5f) / (spc + st + 1e-5f);
  float base = dice + bce / N;
  float boundary = bnd / N;
  float bg = N - fg;
  float fw = fminf(fmaxf(bg / (fg + 1e-7f), 1.0f), 10.0f);
  float fgl = (fw * bfg + bbg) / N;
  if (fg == 0.0f) fgl = 0.0f;

  int tk = task[b];
  float bw = (tk == 0) ? 2.0f : ((tk == 1) ? 3.0f : 5.0f);
  float fw3 = (tk == 0) ? 1.0f : ((tk == 1) ? 1.5f : 3.0f);
  red[b] = base + bw * boundary + fw3 * fgl;  // BASE_W == 1 for all tasks
  __syncthreads();
  if (b == 0) {
    float ssum = 0.0f;
    for (int i = 0; i < NB; ++i) ssum += red[i];
    out[0] = ssum / (float)NB;
  }
}

// ---------------------------------------------------------------------------
extern "C" void kernel_launch(void* const* d_in, const int* in_sizes, int n_in,
                              void* d_out, int out_size, void* d_ws, size_t ws_size,
                              hipStream_t stream) {
  const float* pred = (const float*)d_in[0];
  const float* tgt  = (const float*)d_in[1];
  const int*   task = (const int*)d_in[2];
  float* out = (float*)d_out;

  // Workspace layout: [0,2048)   per-sample accumulators (64 * 8 f32)
  //                   [4096, +2MB) bitpacked target masks
  float* acc = (float*)d_ws;
  unsigned* bits = (unsigned*)((char*)d_ws + 4096);

  k_zero<<<1, 512, 0, stream>>>(acc);
  k_sums<<<NB * 32, 256, 0, stream>>>(pred, tgt, acc, bits);
  k_boundary<<<NB * 64, 256, 0, stream>>>(pred, bits, acc);
  k_final<<<1, NB, 0, stream>>>(acc, task, out);
}
